// SemanticFeatureFusionModule_69750268887276
// MI455X (gfx1250) — matrix operand order repaired and checked by reading, not compiled, not run on hardware
//
#include <hip/hip_runtime.h>
#include <hip/hip_bf16.h>
#include <math.h>

// ---------------------------------------------------------------------------
// Problem constants (match reference)
// ---------------------------------------------------------------------------
#define E_   512
#define H_   8
#define DH_  64
#define FF_  2048
#define NL_  6
#define B_   4
#define NCLS_ 20
#define L2_  40
#define CP_  96
#define C1_  256
#define C2_  128
#define EPS_ 1e-5f

typedef __attribute__((ext_vector_type(16))) __bf16 v16bf;
typedef __attribute__((ext_vector_type(8)))  float  v8f;

union BFrag { v16bf v; unsigned int d[8]; };

__device__ __forceinline__ unsigned int f2bf_hi(float f) {
    unsigned int u = __builtin_bit_cast(unsigned int, f);
    u += 0x7FFFu + ((u >> 16) & 1u);   // round-to-nearest-even
    return u >> 16;
}
__device__ __forceinline__ unsigned int pack_bf2(float a, float b) {
    return f2bf_hi(a) | (f2bf_hi(b) << 16);
}

// ---------------------------------------------------------------------------
// fp32 -> packed bf16 conversion (flat, count must be even)
// ---------------------------------------------------------------------------
__global__ __launch_bounds__(256)
void cvt_bf16(const float* __restrict__ in, unsigned int* __restrict__ out, long count2)
{
    long id = (long)blockIdx.x * 256 + threadIdx.x;
    if (id >= count2) return;
    out[id] = pack_bf2(in[2 * id], in[2 * id + 1]);
}

// ---------------------------------------------------------------------------
// WMMA bf16 GEMM:  Out[M,Nout] = act( X[M,Cin] * W[Nout,Cin]^T + bias )
//   Xbf / Wbf : packed bf16 (dword = 2 elements), row-major, stride Cin
//   Out   : fp32 result (nullable)
//   OutBf : bf16 result (nullable) -- for tensors consumed only by GEMMs
// Block = 256 threads (8 wave32), tile 128(M) x 128(N), K-step 32.
// Each wave owns a 32x64 sub-tile = 2x4 v_wmma_f32_16x16x32_bf16.
// Cin % 32 == 0, Nout % 128 == 0. LDS dword stride 20 => all fragment loads
// are aligned 4-dword (b128) chunks, conflict-free over lanes.
// Staging: CDNA5 async global->LDS DMA (global_load_async_to_lds_b128) with
// DOUBLE-BUFFERED LDS: tile k+1 is in flight while tile k computes; one
// barrier per K-step; wave syncs its own DMA with s_wait_asynccnt.
// ---------------------------------------------------------------------------
__global__ __launch_bounds__(256)
void gemm_bf16(const unsigned int* __restrict__ Xbf, const unsigned int* __restrict__ Wbf,
               const float* __restrict__ bias, float* __restrict__ Out,
               unsigned short* __restrict__ OutBf,
               int M, int Cin, int Nout, int relu)
{
    __shared__ unsigned int XsU[2 * 128 * 20];   // 2 buffers x 128 rows x 16 dwords + pad
    __shared__ unsigned int WsU[2 * 128 * 20];

    const int t    = threadIdx.x;
    const int lane = t & 31;
    const int wv   = t >> 5;          // 0..7
    const int wm   = wv & 3;          // wave row block (4 x 32 rows)
    const int wn   = wv >> 2;         // wave col block (2 x 64 cols)
    const int half = lane >> 4;       // 0: lanes 0-15, 1: lanes 16-31
    const int l16  = lane & 15;

    const int rowBase = blockIdx.x * 128;
    const int colBase = blockIdx.y * 128;
    const int cin2    = Cin >> 1;     // row stride in dwords

    // issue the 4 async b128 copies (2 X + 2 W) for K-dword-offset kd into buffer buf
    auto stage = [&](int buf, int kd) {
        const int bufD = buf * (128 * 20);
        #pragma unroll
        for (int i = 0; i < 2; ++i) {
            int q   = t + i * 256;              // 0..511
            int row = q >> 2, c4 = q & 3;
            int gr  = rowBase + row; if (gr >= M) gr = M - 1;   // clamp: branchless
            const unsigned int* gx = Xbf + (size_t)gr * cin2 + kd + c4 * 4;
            unsigned lx = (unsigned)(size_t)&XsU[bufD + row * 20 + c4 * 4];
            asm volatile("global_load_async_to_lds_b128 %0, %1, off"
                         :: "v"(lx), "v"(gx) : "memory");
            const unsigned int* gw = Wbf + (size_t)(colBase + row) * cin2 + kd + c4 * 4;
            unsigned lw = (unsigned)(size_t)&WsU[bufD + row * 20 + c4 * 4];
            asm volatile("global_load_async_to_lds_b128 %0, %1, off"
                         :: "v"(lw), "v"(gw) : "memory");
        }
    };

    v8f acc[2][4] = {};

    stage(0, 0);                                  // prologue: tile 0 -> buffer 0
    int cur = 0;

    for (int k0 = 0; k0 < Cin; k0 += 32) {
        // my DMA for buf[cur] done; barrier => everyone's done (and all reads of
        // buf[cur^1] from the previous iteration have retired)
        asm volatile("s_wait_asynccnt 0x0" ::: "memory");
        __syncthreads();

        // kick off tile k+1 into the other buffer while we compute this one
        if (k0 + 32 < Cin) stage(cur ^ 1, (k0 + 32) >> 1);
        // prefetch two tiles ahead into L2 (global_prefetch_b8)
        if (k0 + 64 < Cin) {
            int row = t >> 1, c = (t & 1) * 8;
            int gr = rowBase + row; if (gr >= M) gr = M - 1;
            __builtin_prefetch(Xbf + (size_t)gr * cin2 + (k0 >> 1) + 32 + c, 0, 1);
        }

        // ---- fragments (documented wave32 16-bit layouts), b128 LDS reads ----
        const int bufD = cur * (128 * 20);
        BFrag a[2], b[4];
        #pragma unroll
        for (int tm = 0; tm < 2; ++tm) {
            const int rowD = bufD + (wm * 32 + tm * 16 + l16) * 20;
            const int ko2  = half ? 4 : 0;     // K offset 8 halves = 4 dwords
            #pragma unroll
            for (int j = 0; j < 4; ++j) {
                a[tm].d[j]     = XsU[rowD + ko2 + j];       // K =  ko..ko+7
                a[tm].d[j + 4] = XsU[rowD + 8 + ko2 + j];   // K = 16+ko..
            }
        }
        #pragma unroll
        for (int tn = 0; tn < 4; ++tn) {
            const int rowD = bufD + (wn * 64 + tn * 16 + l16) * 20;
            const int ko2  = half ? 4 : 0;     // K offset 8 halves = 4 dwords
            #pragma unroll
            for (int j = 0; j < 4; ++j) {
                b[tn].d[j]     = WsU[rowD + ko2 + j];
                b[tn].d[j + 4] = WsU[rowD + 8 + ko2 + j];
            }
        }

        #pragma unroll
        for (int tm = 0; tm < 2; ++tm)
            #pragma unroll
            for (int tn = 0; tn < 4; ++tn)
                acc[tm][tn] = __builtin_amdgcn_wmma_f32_16x16x32_bf16(
                    false, a[tm].v, false, b[tn].v,
                    (short)0, acc[tm][tn], false, false);

        cur ^= 1;
    }

    // ---- epilogue: bias (+ReLU), fp32 and/or bf16 stores per C/D layout ----
    #pragma unroll
    for (int tm = 0; tm < 2; ++tm)
        #pragma unroll
        for (int tn = 0; tn < 4; ++tn) {
            int gc = colBase + wn * 64 + tn * 16 + l16;
            float bv = bias[gc];
            #pragma unroll
            for (int r = 0; r < 8; ++r) {
                int gr = rowBase + wm * 32 + tm * 16 + half * 8 + r;
                if (gr >= M) continue;
                float v = acc[tm][tn][r] + bv;
                if (relu) v = fmaxf(v, 0.0f);
                if (Out)   Out[(size_t)gr * Nout + gc] = v;
                if (OutBf) OutBf[(size_t)gr * Nout + gc] = (unsigned short)f2bf_hi(v);
            }
        }
}

// ---------------------------------------------------------------------------
// Fused residual + LayerNorm over rows of width E=512.
// out = LN(x + y) * g + b   (y nullable); optional packed-bf16 shadow output.
// Thread t owns adjacent elements {2t, 2t+1}.
// ---------------------------------------------------------------------------
__global__ __launch_bounds__(256)
void add_ln(const float* __restrict__ x, const float* __restrict__ y,
            const float* __restrict__ g, const float* __restrict__ b,
            float* __restrict__ out, unsigned int* __restrict__ out_bf, int rows)
{
    const int row = blockIdx.x;
    if (row >= rows) return;
    const int t = threadIdx.x;
    const size_t base = (size_t)row * E_;
    const int e0 = 2 * t;

    float s0 = x[base + e0]     + (y ? y[base + e0]     : 0.0f);
    float s1 = x[base + e0 + 1] + (y ? y[base + e0 + 1] : 0.0f);

    __shared__ float rs[256], rq[256];
    rs[t] = s0 + s1;
    rq[t] = s0 * s0 + s1 * s1;
    __syncthreads();
    for (int off = 128; off > 0; off >>= 1) {
        if (t < off) { rs[t] += rs[t + off]; rq[t] += rq[t + off]; }
        __syncthreads();
    }
    float mean = rs[0] * (1.0f / E_);
    float var  = rq[0] * (1.0f / E_) - mean * mean;
    float inv  = rsqrtf(var + EPS_);

    float o0 = (s0 - mean) * inv * g[e0]     + b[e0];
    float o1 = (s1 - mean) * inv * g[e0 + 1] + b[e0 + 1];
    *(float2*)(out + base + e0) = make_float2(o0, o1);
    if (out_bf) out_bf[(base >> 1) + t] = pack_bf2(o0, o1);
}

// ---------------------------------------------------------------------------
// Embedding projections: mem[l,b,:] from emb1/emb2 (tiny: 160x512 outputs)
// ---------------------------------------------------------------------------
__global__ __launch_bounds__(256)
void embed_k(const float* __restrict__ e1, const float* __restrict__ e2,
             const float* __restrict__ w1, const float* __restrict__ b1,
             const float* __restrict__ w2, const float* __restrict__ b2,
             float* __restrict__ mem, unsigned short* __restrict__ mem_bf)
{
    int id = blockIdx.x * 256 + threadIdx.x;       // L2*B*E = 81920
    if (id >= L2_ * B_ * E_) return;
    int e  = id & (E_ - 1);
    int lb = id >> 9;
    int b  = lb & (B_ - 1);
    int l  = lb >> 2;
    float acc;
    if (l < NCLS_) {
        acc = b1[e];
        for (int c = 0; c < C1_; ++c)
            acc += e1[(size_t)(b * C1_ + c) * NCLS_ + l] * w1[(size_t)e * C1_ + c];
    } else {
        int l2 = l - NCLS_;
        acc = b2[e];
        for (int c = 0; c < C2_; ++c)
            acc += e2[(size_t)(b * C2_ + c) * NCLS_ + l2] * w2[(size_t)e * C2_ + c];
    }
    mem[id] = acc;
    mem_bf[id] = (unsigned short)f2bf_hi(acc);
}

// ---------------------------------------------------------------------------
// Self-attention over the tiny memory (L2=40, B=4): one thread per (l,b,h)
// qkv: [160, 3E] fp32; out: [160, E] packed bf16 (feeds sa_out GEMM only)
// ---------------------------------------------------------------------------
__global__ __launch_bounds__(256)
void mem_self_attn(const float* __restrict__ qkv, unsigned int* __restrict__ out_bf)
{
    int id = blockIdx.x * 256 + threadIdx.x;       // 40*4*8 = 1280
    if (id >= L2_ * B_ * H_) return;
    int h  = id & (H_ - 1);
    int lb = id >> 3;
    int b  = lb & (B_ - 1);
    int l  = lb >> 2;
    const float scale = 0.125f;                    // DH^-0.5

    const float* q = qkv + (size_t)(l * B_ + b) * (3 * E_) + h * DH_;
    float sc[L2_];
    float mx = -1e30f;
    for (int m = 0; m < L2_; ++m) {
        const float* k = qkv + (size_t)(m * B_ + b) * (3 * E_) + E_ + h * DH_;
        float s = 0.0f;
        for (int d = 0; d < DH_; ++d) s += q[d] * k[d];
        s *= scale;
        sc[m] = s;
        mx = fmaxf(mx, s);
    }
    float sum = 0.0f;
    for (int m = 0; m < L2_; ++m) { sc[m] = __expf(sc[m] - mx); sum += sc[m]; }
    float inv = 1.0f / sum;
    unsigned int* o = out_bf + ((size_t)(l * B_ + b) * E_ + h * DH_) / 2;
    for (int d = 0; d < DH_; d += 2) {
        float a0 = 0.0f, a1 = 0.0f;
        for (int m = 0; m < L2_; ++m) {
            const float* vr = qkv + (size_t)(m * B_ + b) * (3 * E_) + 2 * E_ + h * DH_ + d;
            a0 += sc[m] * vr[0];
            a1 += sc[m] * vr[1];
        }
        o[d >> 1] = pack_bf2(a0 * inv, a1 * inv);
    }
}

// ---------------------------------------------------------------------------
// K/V projections of memory: ck/cv[b, f, l] = sum_c mem[l,b,c]*W[f,c] + bias
// ---------------------------------------------------------------------------
__global__ __launch_bounds__(256)
void kv_proj(const float* __restrict__ mem,
             const float* __restrict__ kw, const float* __restrict__ kb,
             const float* __restrict__ vw, const float* __restrict__ vb,
             float* __restrict__ ck, float* __restrict__ cv)
{
    int id = blockIdx.x * 256 + threadIdx.x;       // B*E*L2 = 81920
    if (id >= B_ * E_ * L2_) return;
    int l = id % L2_;
    int f = (id / L2_) % E_;
    int b = id / (L2_ * E_);
    const float* m = mem + (size_t)(l * B_ + b) * E_;
    float ak = kb[f], av = vb[f];
    for (int c = 0; c < E_; ++c) {
        float mv = m[c];
        ak += mv * kw[(size_t)f * E_ + c];
        av += mv * vw[(size_t)f * E_ + c];
    }
    ck[id] = ak;
    cv[id] = av;
}

// ---------------------------------------------------------------------------
// Per-point cross attention: one thread per (point, head). L2=40 softmax in
// registers; ck/cv (320 KB each) are L2-resident. Output packed bf16 (feeds GEMM).
// ---------------------------------------------------------------------------
__global__ __launch_bounds__(256)
void point_attn(const float* __restrict__ cq, const float* __restrict__ ck,
                const float* __restrict__ cv, const int* __restrict__ bidx,
                unsigned int* __restrict__ of_bf, int n)
{
    int id = blockIdx.x * 256 + threadIdx.x;       // n*H
    if (id >= n * H_) return;
    int h  = id & (H_ - 1);
    int pt = id >> 3;
    int b  = bidx[pt];
    const float scale = 0.125f;

    float sim[L2_];
    #pragma unroll
    for (int l = 0; l < L2_; ++l) sim[l] = 0.0f;

    const float* q   = cq + (size_t)pt * E_ + h * DH_;
    const float* kb_ = ck + ((size_t)b * E_ + h * DH_) * L2_;
    for (int d = 0; d < DH_; ++d) {
        float qd = q[d];
        const float* kr = kb_ + (size_t)d * L2_;
        #pragma unroll
        for (int l = 0; l < L2_; ++l) sim[l] += qd * kr[l];
    }
    float mx = -1e30f;
    #pragma unroll
    for (int l = 0; l < L2_; ++l) { sim[l] *= scale; mx = fmaxf(mx, sim[l]); }
    float sum = 0.0f;
    #pragma unroll
    for (int l = 0; l < L2_; ++l) { sim[l] = __expf(sim[l] - mx); sum += sim[l]; }
    float inv = 1.0f / sum;

    const float* vb_ = cv + ((size_t)b * E_ + h * DH_) * L2_;
    unsigned int* o = of_bf + ((size_t)pt * E_ + h * DH_) / 2;
    for (int d = 0; d < DH_; d += 2) {
        const float* v0 = vb_ + (size_t)d * L2_;
        const float* v1 = v0 + L2_;
        float a0 = 0.0f, a1 = 0.0f;
        #pragma unroll
        for (int l = 0; l < L2_; ++l) { a0 += sim[l] * v0[l]; a1 += sim[l] * v1[l]; }
        o[d >> 1] = pack_bf2(a0 * inv, a1 * inv);
    }
}

// ---------------------------------------------------------------------------
// Host launcher
// ---------------------------------------------------------------------------
extern "C" void kernel_launch(void* const* d_in, const int* in_sizes, int n_in,
                              void* d_out, int out_size, void* d_ws, size_t ws_size,
                              hipStream_t stream)
{
    (void)n_in; (void)out_size; (void)ws_size;

    // ----- inputs (setup_inputs dict order; params dict flattened in order) --
    const float* pt    = (const float*)d_in[0];   // [N, CP]
    const float* emb1  = (const float*)d_in[1];   // [B, C1, NCLS, 1]
    const float* emb2  = (const float*)d_in[2];   // [B, C2, NCLS, 1]
    const int*   bidx  = (const int*)  d_in[3];   // [N]
    // d_in[4] = batch_size scalar (unused; B_ constant)
    int pi = 5;
    const float* wp      = (const float*)d_in[pi++];
    const float* bp      = (const float*)d_in[pi++];
    const float* we1     = (const float*)d_in[pi++];
    const float* be1     = (const float*)d_in[pi++];
    const float* we2     = (const float*)d_in[pi++];
    const float* be2     = (const float*)d_in[pi++];
    const float* sa_in_w = (const float*)d_in[pi++];
    const float* sa_in_b = (const float*)d_in[pi++];
    const float* sa_out_w= (const float*)d_in[pi++];
    const float* sa_out_b= (const float*)d_in[pi++];
    const float* ca_q_w  = (const float*)d_in[pi++];
    const float* ca_q_b  = (const float*)d_in[pi++];
    const float* ca_k_w  = (const float*)d_in[pi++];
    const float* ca_k_b  = (const float*)d_in[pi++];
    const float* ca_v_w  = (const float*)d_in[pi++];
    const float* ca_v_b  = (const float*)d_in[pi++];
    const float* ca_o_w  = (const float*)d_in[pi++];
    const float* ca_o_b  = (const float*)d_in[pi++];
    const float* ff1_w   = (const float*)d_in[pi++];
    const float* ff1_b   = (const float*)d_in[pi++];
    const float* ff2_w   = (const float*)d_in[pi++];
    const float* ff2_b   = (const float*)d_in[pi++];
    const float* ln1_g   = (const float*)d_in[pi++];
    const float* ln1_b   = (const float*)d_in[pi++];
    const float* ln2_g   = (const float*)d_in[pi++];
    const float* ln2_b   = (const float*)d_in[pi++];
    const float* ln3_g   = (const float*)d_in[pi++];
    const float* ln3_b   = (const float*)d_in[pi++];
    const float* lnf_g   = (const float*)d_in[pi++];
    const float* lnf_b   = (const float*)d_in[pi++];

    const int n = in_sizes[0] / CP_;              // number of points
    const int MROWS = L2_ * B_;                   // 160

    // ----- workspace layout ---------------------------------------------------
    float* ws   = (float*)d_ws;
    size_t off  = 0;
    float* tgt  = ws + off; off += (size_t)n * E_;            // [N,E] fp32
    float* bufA = ws + off; off += (size_t)n * E_;            // [N,E] fp32 (cq / gemm out)
    float* mem  = ws + off; off += (size_t)MROWS * E_;        // [160,E] fp32
    float* qkv  = ws + off; off += (size_t)MROWS * 3 * E_;    // [160,3E] (+ sa_out tmp)
    float* ck   = ws + off; off += (size_t)B_ * E_ * L2_;
    float* cv   = ws + off; off += (size_t)B_ * E_ * L2_;

    // packed-bf16 regions (dword = 2 elements); offsets stay 16B-aligned
    unsigned int* bfp = (unsigned int*)(ws + off);
    size_t boff = 0;
    unsigned int* pt_bf   = bfp + boff; boff += (size_t)n * CP_ / 2;
    unsigned int* tgt_bf  = bfp + boff; boff += (size_t)n * E_ / 2;
    unsigned int* h_bf    = bfp + boff; boff += (size_t)n * FF_ / 2;   // ffn hidden; also 'of'
    unsigned int* mem_bf  = bfp + boff; boff += (size_t)MROWS * E_ / 2;
    unsigned int* memT_bf = bfp + boff; boff += (size_t)MROWS * E_ / 2;
    unsigned int* of_bf   = h_bf;                                      // reuse (sequential)

    // bf16 weight pool
    unsigned int* wbf = bfp + boff;
    size_t woff = 0;
    unsigned int* wp_bf  = wbf + woff; woff += (size_t)E_ * CP_ / 2;
    unsigned int* siw_bf = wbf + woff; woff += (size_t)NL_ * 3 * E_ * E_ / 2;
    unsigned int* sow_bf = wbf + woff; woff += (size_t)NL_ * E_ * E_ / 2;
    unsigned int* qw_bf  = wbf + woff; woff += (size_t)NL_ * E_ * E_ / 2;
    unsigned int* ow_bf  = wbf + woff; woff += (size_t)NL_ * E_ * E_ / 2;
    unsigned int* f1w_bf = wbf + woff; woff += (size_t)NL_ * FF_ * E_ / 2;
    unsigned int* f2w_bf = wbf + woff; woff += (size_t)NL_ * E_ * FF_ / 2;

    auto cvt = [&](const float* src, unsigned int* dst, size_t count) {
        long c2 = (long)(count / 2);
        cvt_bf16<<<dim3((unsigned)((c2 + 255) / 256)), dim3(256), 0, stream>>>(src, dst, c2);
    };
    auto gemm = [&](const unsigned int* Xb, const unsigned int* Wb, const float* bias,
                    float* Out, unsigned short* OutBf, int M, int Cin, int Nout, int relu) {
        dim3 grid((M + 127) / 128, Nout / 128);
        gemm_bf16<<<grid, dim3(256), 0, stream>>>(Xb, Wb, bias, Out, OutBf,
                                                  M, Cin, Nout, relu);
    };

    // ----- one-time weight + input conversion to bf16 ------------------------
    cvt(wp,      wp_bf,  (size_t)E_ * CP_);
    cvt(sa_in_w, siw_bf, (size_t)NL_ * 3 * E_ * E_);
    cvt(sa_out_w,sow_bf, (size_t)NL_ * E_ * E_);
    cvt(ca_q_w,  qw_bf,  (size_t)NL_ * E_ * E_);
    cvt(ca_o_w,  ow_bf,  (size_t)NL_ * E_ * E_);
    cvt(ff1_w,   f1w_bf, (size_t)NL_ * FF_ * E_);
    cvt(ff2_w,   f2w_bf, (size_t)NL_ * E_ * FF_);
    cvt(pt,      pt_bf,  (size_t)n * CP_);

    // ----- input projections -------------------------------------------------
    gemm(pt_bf, wp_bf, bp, tgt, (unsigned short*)tgt_bf, n, CP_, E_, 0);
    embed_k<<<(L2_ * B_ * E_) / 256, 256, 0, stream>>>(
        emb1, emb2, we1, be1, we2, be2, mem, (unsigned short*)mem_bf);

    for (int i = 0; i < NL_; ++i) {
        const unsigned int* siw = siw_bf + (size_t)i * 3 * E_ * E_ / 2;
        const unsigned int* sow = sow_bf + (size_t)i * E_ * E_ / 2;
        const unsigned int* qw  = qw_bf  + (size_t)i * E_ * E_ / 2;
        const unsigned int* owt = ow_bf  + (size_t)i * E_ * E_ / 2;
        const unsigned int* f1w = f1w_bf + (size_t)i * FF_ * E_ / 2;
        const unsigned int* f2w = f2w_bf + (size_t)i * E_ * FF_ / 2;
        const float* sib = sa_in_b  + (size_t)i * 3 * E_;
        const float* sob = sa_out_b + (size_t)i * E_;
        const float* qb  = ca_q_b   + (size_t)i * E_;
        const float* kw  = ca_k_w   + (size_t)i * E_ * E_;
        const float* kb2 = ca_k_b   + (size_t)i * E_;
        const float* vw  = ca_v_w   + (size_t)i * E_ * E_;
        const float* vb2 = ca_v_b   + (size_t)i * E_;
        const float* ob  = ca_o_b   + (size_t)i * E_;
        const float* f1b = ff1_b    + (size_t)i * FF_;
        const float* f2b = ff2_b    + (size_t)i * E_;
        const float* g1  = ln1_g + (size_t)i * E_; const float* l1 = ln1_b + (size_t)i * E_;
        const float* g2  = ln2_g + (size_t)i * E_; const float* l2 = ln2_b + (size_t)i * E_;
        const float* g3  = ln3_g + (size_t)i * E_; const float* l3 = ln3_b + (size_t)i * E_;

        // ---- self-attention over memory (tiny) ----
        gemm(mem_bf, siw, sib, qkv, nullptr, MROWS, E_, 3 * E_, 0);
        mem_self_attn<<<(MROWS * H_ + 255) / 256, 256, 0, stream>>>(qkv, memT_bf);
        gemm(memT_bf, sow, sob, qkv /*[160,E] tmp*/, nullptr, MROWS, E_, E_, 0);
        add_ln<<<MROWS, 256, 0, stream>>>(mem, qkv, g1, l1, mem, mem_bf, MROWS);

        // ---- cross-attention ----
        kv_proj<<<(B_ * E_ * L2_) / 256, 256, 0, stream>>>(mem, kw, kb2, vw, vb2, ck, cv);
        gemm(tgt_bf, qw, qb, bufA, nullptr, n, E_, E_, 0);          // cq (fp32)
        point_attn<<<((size_t)n * H_ + 255) / 256, 256, 0, stream>>>(
            bufA, ck, cv, bidx, of_bf, n);                          // of (bf16)
        gemm(of_bf, owt, ob, bufA, nullptr, n, E_, E_, 0);          // o
        add_ln<<<n, 256, 0, stream>>>(tgt, bufA, g2, l2, tgt, tgt_bf, n);

        // ---- FFN ----
        gemm(tgt_bf, f1w, f1b, nullptr, (unsigned short*)h_bf, n, E_, FF_, 1);
        gemm(h_bf, f2w, f2b, bufA, nullptr, n, FF_, E_, 0);
        add_ln<<<n, 256, 0, stream>>>(tgt, bufA, g3, l3, tgt, tgt_bf, n);
    }

    // ----- final LayerNorm -> output ----------------------------------------
    add_ln<<<n, 256, 0, stream>>>(tgt, nullptr, lnf_g, lnf_b, (float*)d_out, nullptr, n);
}
